// RPN_79130477462046
// MI455X (gfx1250) — compile-verified, matrix operand order checked by reading, not run on hardware
//
#include <hip/hip_runtime.h>
#include <math.h>
#include <stdint.h>

typedef __attribute__((ext_vector_type(16))) __bf16 v16bf;
typedef __attribute__((ext_vector_type(8)))  float  v8f;

union Frag32 { uint4 u[2]; v16bf v; };

// ---------------------------------------------------------------------------
// Prep kernel: pack conv weights (O=256, I=256, 3, 3) f32 -> bf16 WMMA B-frag
// layout: [tk = tap*8+kc][nc(16)][lane(32)][j(16)]
//   lane: n = nc*16 + (lane&15), K-half = lane>>4
//   j:    k = kc*32 + half*16 + j   (k = input channel within tap)
// ---------------------------------------------------------------------------
__global__ void pack_conv_w(const float* __restrict__ cw, __bf16* __restrict__ wp) {
  int f = blockIdx.x * 256 + threadIdx.x;
  if (f >= 36864) return;            // 9*8*16*32
  int lane = f & 31;
  int nc   = (f >> 5) & 15;
  int kc   = (f >> 9) & 7;
  int tap  = f >> 12;
  int half = lane >> 4, l16 = lane & 15;
  int n = nc * 16 + l16;
  #pragma unroll
  for (int j = 0; j < 16; ++j) {
    int k = kc * 32 + half * 16 + j;
    wp[(size_t)f * 16 + j] = (__bf16)cw[((size_t)n * 256 + k) * 9 + tap];
  }
}

// Pack head weights: rows 0-1 = cls_w, rows 2-46 = off_w, row 47 = 0 (pad).
// layout: [(kc*3+nc)][lane(32)][j(16)]
__global__ void pack_head_w(const float* __restrict__ clsw,
                            const float* __restrict__ offw,
                            __bf16* __restrict__ wp) {
  int f = blockIdx.x * 256 + threadIdx.x;
  if (f >= 768) return;              // 8*3*32
  int lane = f & 31;
  int t2 = f >> 5;
  int nc = t2 % 3;
  int kc = t2 / 3;
  int half = lane >> 4, l16 = lane & 15;
  int n = nc * 16 + l16;
  #pragma unroll
  for (int j = 0; j < 16; ++j) {
    int k = kc * 32 + half * 16 + j;
    float v = 0.f;
    if (n < 2)       v = clsw[n * 256 + k];
    else if (n < 47) v = offw[(n - 2) * 256 + k];
    wp[f * 16 + j] = (__bf16)v;
  }
}

// NCHW f32 -> NHWC bf16 (per level, both batches)
__global__ void cvt_nchw_to_nhwc_bf16(const float* __restrict__ src,
                                      __bf16* __restrict__ dst, int n) {
  long i = (long)blockIdx.x * 256 + threadIdx.x;
  long total = (long)2 * 256 * n;
  if (i >= total) return;
  int p = (int)(i % n);
  int c = (int)((i / n) & 255);
  int b = (int)(i / ((long)n * 256));
  dst[((long)(b * n + p)) * 256 + c] = (__bf16)src[i];
}

// ---------------------------------------------------------------------------
// Fused per-level RPN kernel.  Block = 256 thr = 8 waves, owns 32 pixels.
//   phase 1: implicit-GEMM 3x3 conv via v_wmma_f32_16x16x32_bf16
//            A-tile staged global->LDS with GLOBAL_LOAD_ASYNC_TO_LDS_B128
//            wave tile: 2 M-subtiles x 2 N-subtiles (B-fragments reused)
//   phase 2: bias+relu -> bf16 t-tile in LDS
//   phase 3: 32x48 head GEMM (cls + off) via WMMA (waves 0..5)
//   phase 4: gumbel-softmax bbox epilogue, scatter to output layout
// ---------------------------------------------------------------------------
#define LDS_STRIDE 264   // 256 + 8 pad: 528B row -> 4-dword bank skew, 16B aligned
#define MTILE 32

__global__ __launch_bounds__(256)
void rpn_level_kernel(const __bf16* __restrict__ featN,   // [n][256] this (l,b)
                      const __bf16* __restrict__ wPack,
                      const __bf16* __restrict__ w2Pack,
                      const float* __restrict__ conv_b,
                      const float* __restrict__ cls_b,
                      const float* __restrict__ off_b,
                      const float* __restrict__ eps,      // [20][n] this (l,b)
                      const float* __restrict__ gum,      // [5][n]  this (l,b)
                      float* __restrict__ out,            // + b*TOT + level off
                      int h, int w, int n)
{
  __shared__ __align__(16) __bf16 Alds[MTILE * LDS_STRIDE];
  __shared__ __align__(16) __bf16 Tlds[MTILE * LDS_STRIDE];
  __shared__ __align__(16) float  Olds[MTILE * 48];

  const int tid  = threadIdx.x;
  const int p0   = blockIdx.x * MTILE;
  const int wid  = tid >> 5;
  const int lane = tid & 31;
  const int half = lane >> 4;
  const int l16  = lane & 15;
  const int nc0  = wid * 2, nc1 = wid * 2 + 1;

  const unsigned long long gbase = (unsigned long long)(uintptr_t)featN;

  // ---- tap-invariant staging state (4 units of 8 channels per thread) ----
  int     sy[4], sx[4];
  bool    pv[4];
  unsigned ldsoff[4];     // LDS byte offset of destination
  unsigned vbase[4];      // global byte offset of (y, x, cu*8) in featN
  __bf16*  dptr[4];       // LDS pointer (zero-fill fallback)
  #pragma unroll
  for (int it = 0; it < 4; ++it) {
    int u  = tid + it * 256;          // 1024 units
    int m  = u >> 5;
    int cu = u & 31;
    int p  = p0 + m;
    int y  = p / w, x = p - y * w;
    sy[it] = y; sx[it] = x; pv[it] = (p < n);
    dptr[it]   = Alds + m * LDS_STRIDE + cu * 8;
    ldsoff[it] = (unsigned)(uintptr_t)(void*)dptr[it];
    vbase[it]  = (((unsigned)(y * w + x)) * 256u + (unsigned)cu * 8u) * 2u;
  }

  v8f acc00 = {0.f,0.f,0.f,0.f,0.f,0.f,0.f,0.f};
  v8f acc01 = acc00, acc10 = acc00, acc11 = acc00;

  for (int tap = 0; tap < 9; ++tap) {
    const int dy = tap / 3 - 1, dx = tap % 3 - 1;
    const int delta = (dy * w + dx) * 512;    // uniform byte shift per tap
    if (tap < 8)   // warm caches with next tap's weight fragments
      __builtin_prefetch(wPack + (size_t)(tap + 1) * 65536, 0, 3);
    __syncthreads();   // previous tap's A reads complete before overwrite
    // stage shifted 32x256 bf16 A-tile (zero padded) -> LDS via async DMA
    #pragma unroll
    for (int it = 0; it < 4; ++it) {
      int ys = sy[it] + dy;
      int xs = sx[it] + dx;
      if (pv[it] && ys >= 0 && ys < h && xs >= 0 && xs < w) {
        unsigned voff = vbase[it] + (unsigned)delta;
        asm volatile("global_load_async_to_lds_b128 %0, %1, %2"
                     :: "v"(ldsoff[it]), "v"(voff), "s"(gbase) : "memory");
      } else {
        uint4 z = {0u, 0u, 0u, 0u};
        *(uint4*)dptr[it] = z;        // zero padding (SAME conv + tile tail)
      }
    }
    asm volatile("s_wait_asynccnt 0" ::: "memory");
    __syncthreads();
    // per-tap fragment bases: inner kc offsets become 24-bit immediates
    const __bf16* at  = Alds + l16 * LDS_STRIDE + half * 16;
    const __bf16* bt0 = wPack + (size_t)tap * 65536 + ((size_t)nc0 * 32 + lane) * 16;
    const __bf16* bt1 = wPack + (size_t)tap * 65536 + ((size_t)nc1 * 32 + lane) * 16;
    #pragma unroll
    for (int kc = 0; kc < 8; ++kc) {
      Frag32 a0, a1, b0, b1;
      const __bf16* ap0 = at + kc * 32;
      const __bf16* ap1 = ap0 + 16 * LDS_STRIDE;
      a0.u[0] = *(const uint4*)(ap0);
      a0.u[1] = *(const uint4*)(ap0 + 8);
      a1.u[0] = *(const uint4*)(ap1);
      a1.u[1] = *(const uint4*)(ap1 + 8);
      const __bf16* bp0 = bt0 + kc * 8192;
      const __bf16* bp1 = bt1 + kc * 8192;
      b0.u[0] = *(const uint4*)(bp0);
      b0.u[1] = *(const uint4*)(bp0 + 8);
      b1.u[0] = *(const uint4*)(bp1);
      b1.u[1] = *(const uint4*)(bp1 + 8);
      acc00 = __builtin_amdgcn_wmma_f32_16x16x32_bf16(false, a0.v, false, b0.v,
                                                      (short)0, acc00, false, false);
      acc01 = __builtin_amdgcn_wmma_f32_16x16x32_bf16(false, a0.v, false, b1.v,
                                                      (short)0, acc01, false, false);
      acc10 = __builtin_amdgcn_wmma_f32_16x16x32_bf16(false, a1.v, false, b0.v,
                                                      (short)0, acc10, false, false);
      acc11 = __builtin_amdgcn_wmma_f32_16x16x32_bf16(false, a1.v, false, b1.v,
                                                      (short)0, acc11, false, false);
    }
  }

  // phase 2: bias + relu, write bf16 t-tile (32 x 256)
  {
    const int o0 = wid * 32 + l16;
    const int o1 = o0 + 16;
    const float bia0 = conv_b[o0];
    const float bia1 = conv_b[o1];
    #pragma unroll
    for (int r = 0; r < 8; ++r) {
      int m0 = r + half * 8;
      int m1 = m0 + 16;
      Tlds[m0 * LDS_STRIDE + o0] = (__bf16)fmaxf(acc00[r] + bia0, 0.f);
      Tlds[m0 * LDS_STRIDE + o1] = (__bf16)fmaxf(acc01[r] + bia1, 0.f);
      Tlds[m1 * LDS_STRIDE + o0] = (__bf16)fmaxf(acc10[r] + bia0, 0.f);
      Tlds[m1 * LDS_STRIDE + o1] = (__bf16)fmaxf(acc11[r] + bia1, 0.f);
    }
  }
  __syncthreads();

  // phase 3: head GEMM 32x48, K=256 (waves 0..5: wid = msub*3 + nc)
  if (wid < 6) {
    const int nc   = wid % 3;
    const int msub = wid / 3;
    v8f acc2 = {0.f,0.f,0.f,0.f,0.f,0.f,0.f,0.f};
    const __bf16* at = Tlds + (msub * 16 + l16) * LDS_STRIDE + half * 16;
    const __bf16* bt = w2Pack + ((size_t)nc * 32 + lane) * 16;
    #pragma unroll
    for (int kc = 0; kc < 8; ++kc) {
      Frag32 a, b;
      const __bf16* ap = at + kc * 32;
      a.u[0] = *(const uint4*)(ap);
      a.u[1] = *(const uint4*)(ap + 8);
      const __bf16* bp = bt + (size_t)kc * 3 * 512;   // (kc*3+nc)*32*16 elems
      b.u[0] = *(const uint4*)(bp);
      b.u[1] = *(const uint4*)(bp + 8);
      acc2 = __builtin_amdgcn_wmma_f32_16x16x32_bf16(false, a.v, false, b.v,
                                                     (short)0, acc2, false, false);
    }
    #pragma unroll
    for (int r = 0; r < 8; ++r) {
      int m = msub * 16 + r + half * 8;
      Olds[m * 48 + nc * 16 + l16] = acc2[r];
    }
  }
  __syncthreads();

  // phase 4: per-pixel epilogue (one lane per pixel, wave 0)
  if (tid < MTILE) {
    int p = p0 + tid;
    if (p < n) {
      float v[48];
      #pragma unroll
      for (int i = 0; i < 48; ++i) v[i] = Olds[tid * 48 + i];
      float cls0 = v[0] + cls_b[0];
      float cls1 = v[1] + cls_b[1];
      float offv[45];
      #pragma unroll
      for (int i = 0; i < 45; ++i) offv[i] = v[2 + i] + off_b[i];
      // gumbel softmax over K=5, temperature 0.1
      float s[5];
      float mx = -1e30f;
      #pragma unroll
      for (int k = 0; k < 5; ++k) {
        float uk = gum[k * n + p];
        float g  = -logf(-logf(uk + 1e-10f) + 1e-10f);
        s[k] = (g + offv[40 + k]) * 10.0f;
        mx = fmaxf(mx, s[k]);
      }
      float ssum = 0.f;
      #pragma unroll
      for (int k = 0; k < 5; ++k) { s[k] = expf(s[k] - mx); ssum += s[k]; }
      float inv = 1.0f / ssum;
      float bbox[4] = {0.f, 0.f, 0.f, 0.f};
      #pragma unroll
      for (int k = 0; k < 5; ++k) {
        float cw = s[k] * inv;
        #pragma unroll
        for (int j = 0; j < 4; ++j) {
          int kj = k * 4 + j;
          float smp = offv[kj] + expf(offv[20 + kj]) * eps[kj * n + p];
          bbox[j] += cw * smp;
        }
      }
      out[p * 2 + 0] = cls0;
      out[p * 2 + 1] = cls1;
      float* ob = out + 2 * n;
      #pragma unroll
      for (int j = 0; j < 4; ++j) ob[p * 4 + j] = bbox[j];
    }
  }
}

// ---------------------------------------------------------------------------
extern "C" void kernel_launch(void* const* d_in, const int* in_sizes, int n_in,
                              void* d_out, int out_size, void* d_ws, size_t ws_size,
                              hipStream_t stream) {
  (void)in_sizes; (void)n_in; (void)out_size; (void)ws_size;

  static const int H[5] = {200, 100, 50, 25, 13};
  static const int W[5] = {336, 168, 84, 42, 21};
  static const int N[5] = {67200, 16800, 4200, 1050, 273};

  const float* feat[5]; const float* eps[5]; const float* gum[5];
  for (int l = 0; l < 5; ++l) {
    feat[l] = (const float*)d_in[3 * l + 0];
    eps[l]  = (const float*)d_in[3 * l + 1];
    gum[l]  = (const float*)d_in[3 * l + 2];
  }
  const float* conv_w = (const float*)d_in[15];
  const float* conv_b = (const float*)d_in[16];
  const float* cls_w  = (const float*)d_in[17];
  const float* cls_b  = (const float*)d_in[18];
  const float* off_w  = (const float*)d_in[19];
  const float* off_b  = (const float*)d_in[20];
  float* out = (float*)d_out;

  char* ws = (char*)d_ws;
  __bf16* wPack  = (__bf16*)(ws);                 // 36864*16*2 = 1,179,648 B
  __bf16* w2Pack = (__bf16*)(ws + 1179648);       // 768*16*2   =    24,576 B
  __bf16* featN  = (__bf16*)(ws + 1204224);       // 45,835,776 elems bf16

  pack_conv_w<<<144, 256, 0, stream>>>(conv_w, wPack);
  pack_head_w<<<3, 256, 0, stream>>>(cls_w, off_w, w2Pack);

  long featOffE[5]; long accE = 0;
  for (int l = 0; l < 5; ++l) { featOffE[l] = accE; accE += 2L * 256 * N[l]; }

  for (int l = 0; l < 5; ++l) {
    long total = 2L * 256 * N[l];
    int blocks = (int)((total + 255) / 256);
    cvt_nchw_to_nhwc_bf16<<<blocks, 256, 0, stream>>>(feat[l],
                                                      featN + featOffE[l], N[l]);
  }

  const long TOT = 537138;  // per-batch output length
  long Loff = 0;
  for (int l = 0; l < 5; ++l) {
    int blocks = (N[l] + MTILE - 1) / MTILE;
    for (int b = 0; b < 2; ++b) {
      rpn_level_kernel<<<blocks, 256, 0, stream>>>(
          featN + featOffE[l] + (long)b * N[l] * 256,
          wPack, w2Pack, conv_b, cls_b, off_b,
          eps[l] + (long)b * 20 * N[l],
          gum[l] + (long)b * 5 * N[l],
          out + (long)b * TOT + Loff,
          H[l], W[l], N[l]);
    }
    Loff += 6L * N[l];
  }
}
